// GIN_Encoder_75428215652549
// MI455X (gfx1250) — compile-verified, matrix operand order checked by reading, not run on hardware
//
#include <hip/hip_runtime.h>
#include <math.h>

// ---------------------------------------------------------------------------
// GIN encoder for MI455X (gfx1250, wave32, WMMA).
// Memory-bound problem (~7 GB traffic vs 78 GFLOP), so everything is fused:
// GEMM (bf16 WMMA, f32 accum) + bias + LN + ReLU (+ 2nd LN + residual) in one
// kernel, weights pre-packed into per-lane WMMA B-fragment order (L2-resident).
// ---------------------------------------------------------------------------

#define DIMD 128
#define LDS_PITCH 136  // bf16 elements per row in LDS A tile (272B, breaks 64-bank alignment)

typedef __attribute__((ext_vector_type(16))) __bf16 v16bf;
typedef __attribute__((ext_vector_type(8)))  float  v8f;

__device__ __forceinline__ unsigned short f32_to_bf16(float f) {
    unsigned int u = __float_as_uint(f);
    unsigned int r = u + 0x7FFFu + ((u >> 16) & 1u);   // round-to-nearest-even
    return (unsigned short)(r >> 16);
}

// Reduce a value across the 16 lanes of this lane's half-wave (wave32).
__device__ __forceinline__ float half_wave_sum(float s) {
    s += __shfl_xor(s, 1, 32);
    s += __shfl_xor(s, 2, 32);
    s += __shfl_xor(s, 4, 32);
    s += __shfl_xor(s, 8, 32);
    return s;
}

// LayerNorm + ReLU over a 128-wide row held as 8 values/lane across 16 lanes.
// m = lane&15 selects which 8 columns this lane owns (cols m*8 .. m*8+7).
__device__ __forceinline__ void row_ln_relu(float x[8], int m,
                                            const float* __restrict__ g,
                                            const float* __restrict__ b) {
    float s = 0.f;
#pragma unroll
    for (int c = 0; c < 8; ++c) s += x[c];
    s = half_wave_sum(s);
    float mean = s * (1.0f / 128.0f);
    float vs = 0.f;
#pragma unroll
    for (int c = 0; c < 8; ++c) { float d = x[c] - mean; vs += d * d; }
    vs = half_wave_sum(vs);
    float inv = rsqrtf(vs * (1.0f / 128.0f) + 1e-5f);
#pragma unroll
    for (int c = 0; c < 8; ++c) {
        int col = m * 8 + c;
        float v = (x[c] - mean) * inv * g[col] + b[col];
        x[c] = fmaxf(v, 0.f);
    }
}

// ---------------------------------------------------------------------------
// Pack W1/W2 (f32, [L][128][128], z@W so element (k,n) at k*128+n) into bf16
// WMMA B-fragment order: [mat][layer][ntile(8)][kstep(4)][lane(32)][comp(16)].
// Per-lane data is 32B contiguous -> coalesced b128 loads in the GEMM.
// ---------------------------------------------------------------------------
__global__ void pack_weights_kernel(const float* __restrict__ W1,
                                    const float* __restrict__ W2,
                                    unsigned short* __restrict__ out,
                                    int Lnum) {
    int idx = blockIdx.x * blockDim.x + threadIdx.x;
    int total = 2 * Lnum * 16384;
    if (idx >= total) return;
    int j    =  idx        & 15;
    int lane = (idx >> 4)  & 31;
    int ks   = (idx >> 9)  & 3;
    int tile = (idx >> 11) & 7;
    int l    = (idx >> 14) % Lnum;
    int mat  =  idx / (16384 * Lnum);
    // ISA 16-bit B layout: lanes 0-15 hold K 0-7 & 16-23, lanes 16-31 get +8.
    int k = ks * 32 + ((j < 8) ? j : j + 8) + 8 * (lane >> 4);
    int n = tile * 16 + (lane & 15);
    const float* W = mat ? W2 : W1;
    out[idx] = f32_to_bf16(W[((size_t)l * DIMD + k) * DIMD + n]);
}

// ---------------------------------------------------------------------------
// LayerNorm over rows of [n,128]; one wave per row, 4 floats per lane.
// ---------------------------------------------------------------------------
__global__ __launch_bounds__(256) void ln_rows_kernel(const float* __restrict__ in,
                                                      const float* __restrict__ g,
                                                      const float* __restrict__ b,
                                                      float* __restrict__ out, int n) {
    int row = blockIdx.x * (blockDim.x >> 5) + (threadIdx.x >> 5);
    if (row >= n) return;
    int lane = threadIdx.x & 31;
    float4 v = ((const float4*)(in + (size_t)row * DIMD))[lane];
    float s = v.x + v.y + v.z + v.w;
#pragma unroll
    for (int msk = 1; msk <= 16; msk <<= 1) s += __shfl_xor(s, msk, 32);
    float mean = s * (1.0f / 128.0f);
    float dx = v.x - mean, dy = v.y - mean, dz = v.z - mean, dw = v.w - mean;
    float vs = dx * dx + dy * dy + dz * dz + dw * dw;
#pragma unroll
    for (int msk = 1; msk <= 16; msk <<= 1) vs += __shfl_xor(vs, msk, 32);
    float inv = rsqrtf(vs * (1.0f / 128.0f) + 1e-5f);
    int c = lane * 4;
    float4 o;
    o.x = dx * inv * g[c + 0] + b[c + 0];
    o.y = dy * inv * g[c + 1] + b[c + 1];
    o.z = dz * inv * g[c + 2] + b[c + 2];
    o.w = dw * inv * g[c + 3] + b[c + 3];
    ((float4*)(out + (size_t)row * DIMD))[lane] = o;
}

__global__ __launch_bounds__(256) void copy_kernel(const float* __restrict__ in,
                                                   float* __restrict__ out, size_t n4) {
    size_t i = (size_t)blockIdx.x * blockDim.x + threadIdx.x;
    if (i < n4) ((float4*)out)[i] = ((const float4*)in)[i];
}

// z[dst[e]] += h[src[e]]  — one wave per edge, float4 per lane, f32 atomics.
__global__ __launch_bounds__(256) void scatter_add_kernel(const float* __restrict__ h,
                                                          const int* __restrict__ src,
                                                          const int* __restrict__ dst,
                                                          float* __restrict__ z, int E) {
    int e = blockIdx.x * (blockDim.x >> 5) + (threadIdx.x >> 5);
    if (e >= E) return;
    int lane = threadIdx.x & 31;
    int s = src[e], d = dst[e];
    float4 v = ((const float4*)(h + (size_t)s * DIMD))[lane];
    float* zp = z + (size_t)d * DIMD + lane * 4;
    atomicAdd(zp + 0, v.x);
    atomicAdd(zp + 1, v.y);
    atomicAdd(zp + 2, v.z);
    atomicAdd(zp + 3, v.w);
}

// ---------------------------------------------------------------------------
// Fused GEMM + bias + LN + ReLU [+ LN2 + ReLU + residual].
// Block = 256 threads = 8 waves; block computes 16 rows x 128 cols.
// Wave w owns output column tile [16w,16w+16); K-loop: 4x wmma_f32_16x16x32_bf16.
// ---------------------------------------------------------------------------
__global__ __launch_bounds__(256) void gemm_ln_kernel(
    const float* __restrict__ A,            // [nrows,128] activations (f32)
    const unsigned short* __restrict__ Bp,  // packed bf16 weights [8][4][32][16]
    const float* __restrict__ bias,         // [128]
    const float* __restrict__ g1, const float* __restrict__ b1v,
    const float* __restrict__ g2, const float* __restrict__ b2v,  // 2nd LN or null
    const float* __restrict__ residual,     // [nrows,128] or null
    float* __restrict__ out, int nrows, int two_ln) {

    __shared__ unsigned short As[16 * LDS_PITCH];
    __shared__ float Cs[16 * DIMD];

    const int row0 = blockIdx.x * 16;
    const int tid  = threadIdx.x;
    const int wave = tid >> 5;
    const int lane = tid & 31;
    const int m    = lane & 15;
    const int half = lane >> 4;
    const int n0   = wave * 16;

    // Stage A tile (16x128) into LDS as bf16.
    for (int idx = tid; idx < 16 * DIMD; idx += 256) {
        int r = idx >> 7, c = idx & 127;
        int gr = row0 + r;
        float f = (gr < nrows) ? A[(size_t)gr * DIMD + c] : 0.f;
        As[r * LDS_PITCH + c] = f32_to_bf16(f);
    }
    __syncthreads();

    v8f acc = {0.f, 0.f, 0.f, 0.f, 0.f, 0.f, 0.f, 0.f};
#pragma unroll
    for (int kk = 0; kk < DIMD; kk += 32) {
        union { unsigned short u[16]; v16bf v; } af;
#pragma unroll
        for (int j = 0; j < 16; ++j) {
            // ISA 16-bit A layout: lanes 0-15 K 0-7/16-23; lanes 16-31 +8.
            int k = kk + ((j < 8) ? j : j + 8) + 8 * half;
            af.u[j] = As[m * LDS_PITCH + k];
        }
        union { unsigned short u[16]; v16bf v; } bf;
        const unsigned short* bp = Bp + ((size_t)(wave * 4 + (kk >> 5)) << 9) + lane * 16;
#pragma unroll
        for (int j = 0; j < 16; ++j) bf.u[j] = bp[j];
        acc = __builtin_amdgcn_wmma_f32_16x16x32_bf16(
            false, af.v, false, bf.v, (short)0, acc, false, false);
    }

    // Spill 16x16 f32 tile to LDS: lane l comp j -> row j+8*half, col n0+(l&15).
#pragma unroll
    for (int j = 0; j < 8; ++j)
        Cs[(j + 8 * half) * DIMD + n0 + m] = acc[j];
    __syncthreads();

    // Epilogue: wave w handles rows 2w (lanes 0-15) and 2w+1 (lanes 16-31);
    // each lane owns 8 columns -> full row visible to its 16-lane group.
    int r    = 2 * wave + half;
    int grow = row0 + r;
    float x[8];
#pragma unroll
    for (int c = 0; c < 8; ++c)
        x[c] = Cs[r * DIMD + m * 8 + c] + bias[m * 8 + c];

    row_ln_relu(x, m, g1, b1v);          // LN + ReLU
    if (two_ln)
        row_ln_relu(x, m, g2, b2v);      // second LN + ReLU (GIN block norm)

    if (grow < nrows) {
        if (residual) {
#pragma unroll
            for (int c = 0; c < 8; ++c)
                x[c] += residual[(size_t)grow * DIMD + m * 8 + c];
        }
#pragma unroll
        for (int c = 0; c < 8; ++c)
            out[(size_t)grow * DIMD + m * 8 + c] = x[c];
    }
}

// ---------------------------------------------------------------------------
// Pooling
// ---------------------------------------------------------------------------
__global__ void zero_kernel(float* __restrict__ p, size_t n) {
    size_t i = (size_t)blockIdx.x * blockDim.x + threadIdx.x;
    if (i < n) p[i] = 0.f;
}

__global__ __launch_bounds__(256) void pool_scatter_kernel(const float* __restrict__ h,
                                                           const int* __restrict__ gids,
                                                           float* __restrict__ sums,
                                                           float* __restrict__ counts, int n) {
    int node = blockIdx.x * (blockDim.x >> 5) + (threadIdx.x >> 5);
    if (node >= n) return;
    int lane = threadIdx.x & 31;
    int g = gids[node];
    float4 v = ((const float4*)(h + (size_t)node * DIMD))[lane];
    float* sp = sums + (size_t)g * DIMD + lane * 4;
    atomicAdd(sp + 0, v.x);
    atomicAdd(sp + 1, v.y);
    atomicAdd(sp + 2, v.z);
    atomicAdd(sp + 3, v.w);
    if (lane == 0) atomicAdd(counts + g, 1.f);
}

__global__ void pool_div_kernel(float* __restrict__ sums,
                                const float* __restrict__ counts, size_t n) {
    size_t i = (size_t)blockIdx.x * blockDim.x + threadIdx.x;
    if (i < n) sums[i] /= fmaxf(counts[i >> 7], 1.f);
}

// ---------------------------------------------------------------------------
// Host launcher
// ---------------------------------------------------------------------------
extern "C" void kernel_launch(void* const* d_in, const int* in_sizes, int n_in,
                              void* d_out, int out_size, void* d_ws, size_t ws_size,
                              hipStream_t stream) {
    const float* h      = (const float*)d_in[0];
    const int*   src    = (const int*)d_in[1];
    const int*   dst    = (const int*)d_in[2];
    const int*   gids   = (const int*)d_in[3];
    const float* iln_g  = (const float*)d_in[4];
    const float* iln_b  = (const float*)d_in[5];
    const float* W1     = (const float*)d_in[6];
    const float* b1     = (const float*)d_in[7];
    const float* ln1g   = (const float*)d_in[8];
    const float* ln1b   = (const float*)d_in[9];
    const float* W2     = (const float*)d_in[10];
    const float* b2     = (const float*)d_in[11];
    const float* ln2g   = (const float*)d_in[12];
    const float* ln2b   = (const float*)d_in[13];
    const float* normg  = (const float*)d_in[14];
    const float* normb  = (const float*)d_in[15];

    const int N    = in_sizes[0] / DIMD;
    const int E    = in_sizes[1];
    const int Lnum = in_sizes[6] / (DIMD * DIMD);
    const int G    = out_size / DIMD;
    float* out = (float*)d_out;

    // Workspace layout: 3 N x 128 f32 buffers + counts + packed bf16 weights.
    size_t ndf = (size_t)N * DIMD;
    float* bufA   = (float*)d_ws;
    float* bufB   = bufA + ndf;
    float* bufC   = bufB + ndf;
    float* counts = bufC + ndf;
    unsigned short* wpack = (unsigned short*)(counts + G);

    // 1. Pack weights to bf16 WMMA fragment order (L2-resident, 512 KB).
    int totalW = 2 * Lnum * 16384;
    pack_weights_kernel<<<(totalW + 255) / 256, 256, 0, stream>>>(W1, W2, wpack, Lnum);

    // 2. Input LayerNorm.
    ln_rows_kernel<<<(N + 7) / 8, 256, 0, stream>>>(h, iln_g, iln_b, bufA, N);

    float* hcur = bufA;   // current h (also residual source)
    float* zbuf = bufB;   // z = h + agg, reused as next h
    float* ybuf = bufC;   // GEMM1 output

    size_t n4 = ndf / 4;
    for (int l = 0; l < Lnum; ++l) {
        // z = h; z += scatter(h[src] -> dst)
        copy_kernel<<<(int)((n4 + 255) / 256), 256, 0, stream>>>(hcur, zbuf, n4);
        scatter_add_kernel<<<(E + 7) / 8, 256, 0, stream>>>(hcur, src, dst, zbuf, E);

        // y = relu(LN(z @ W1 + b1))
        gemm_ln_kernel<<<(N + 15) / 16, 256, 0, stream>>>(
            zbuf, wpack + (size_t)(0 * Lnum + l) * 16384, b1 + l * DIMD,
            ln1g + l * DIMD, ln1b + l * DIMD, nullptr, nullptr, nullptr,
            ybuf, N, 0);

        // h' = relu(LN(relu(LN(y @ W2 + b2)))) [+ residual]; overwrite zbuf.
        gemm_ln_kernel<<<(N + 15) / 16, 256, 0, stream>>>(
            ybuf, wpack + (size_t)(1 * Lnum + l) * 16384, b2 + l * DIMD,
            ln2g + l * DIMD, ln2b + l * DIMD, normg + l * DIMD, normb + l * DIMD,
            (l > 0) ? hcur : nullptr, zbuf, N, 1);

        float* t = hcur; hcur = zbuf; zbuf = t;
    }

    // 3. Mean pooling per graph.
    size_t gn = (size_t)G * DIMD;
    zero_kernel<<<(int)((gn + 255) / 256), 256, 0, stream>>>(out, gn);
    zero_kernel<<<(G + 255) / 256, 256, 0, stream>>>(counts, (size_t)G);
    pool_scatter_kernel<<<(N + 7) / 8, 256, 0, stream>>>(hcur, gids, out, counts, N);
    pool_div_kernel<<<(int)((gn + 255) / 256), 256, 0, stream>>>(out, counts, gn);
}
